// DeformableCrossAttention_58583353917950
// MI455X (gfx1250) — matrix operand order repaired and checked by reading, not compiled, not run on hardware
//
#include <hip/hip_runtime.h>
#include <hip/hip_bf16.h>

typedef __attribute__((ext_vector_type(16))) _Float16 v16h;
typedef __attribute__((ext_vector_type(8)))  _Float16 v8h;
typedef __attribute__((ext_vector_type(8)))  float    v8f;

#define LDA 40   // f16 elems per LDS row (32 + 8 pad): 80B row stride, 16B aligned
#define LDB 40

#define WMMA_F16(a, b, c) __builtin_amdgcn_wmma_f32_16x16x32_f16( \
    false, (a), false, (b), (short)0, (c), false, false)

// ---------------------------------------------------------------------------
// Generic GEMM: C[m][n] = sum_k A[m][k] * W[n][k] + bias[n]
// A: (M,K) f32 row-major; W: (N,K) f32 row-major; C: (M,N) f32.
// f16 WMMA 16x16x32, f32 accumulation. 256 thr = 8 waves in 4(M)x2(N);
// block tile 128x64, wave tile 32x32 (4 WMMA / K-step, register reuse of
// 2 A-frags x 2 B-frags). Double-buffered LDS: one barrier per K-step,
// next tile's global loads issued before the compute phase.
// K must be a multiple of 32; M,N guarded.
// ---------------------------------------------------------------------------
__global__ void __launch_bounds__(256)
gemm_f16wmma_bias(const float* __restrict__ A, const float* __restrict__ W,
                  const float* __restrict__ bias, float* __restrict__ C,
                  int M, int N, int K)
{
    __shared__ __align__(16) _Float16 As[2][128 * LDA];
    __shared__ __align__(16) _Float16 Bs[2][64 * LDB];

    const int tid  = threadIdx.x;
    const int wave = tid >> 5;
    const int lane = tid & 31;
    const int wm   = wave >> 1;          // 0..3 : 32-row sub-tile
    const int wn   = wave & 1;           // 0..1 : 32-col sub-tile
    const int m0   = blockIdx.x * 128;
    const int n0   = blockIdx.y * 64;
    const int g    = lane >> 4;          // half-wave id (0/1)
    const int mn   = lane & 15;

    // staging assignments
    const int ar = tid >> 1;             // 0..127
    const int ac = (tid & 1) * 16;       // 0,16
    const int br = tid >> 2;             // 0..63
    const int bc = (tid & 3) * 8;        // 0,8,16,24
    const bool aval = (m0 + ar) < M;
    const bool bval = (n0 + br) < N;

    float areg[16];
    float breg[8];

    auto load_regs = [&](int k0) {
        const float* asrc = A + (size_t)(m0 + ar) * K + (k0 + ac);
        #pragma unroll
        for (int u = 0; u < 16; u += 4) {
            float4 t = aval ? *(const float4*)(asrc + u) : make_float4(0.f, 0.f, 0.f, 0.f);
            areg[u + 0] = t.x; areg[u + 1] = t.y; areg[u + 2] = t.z; areg[u + 3] = t.w;
        }
        const float* bsrc = W + (size_t)(n0 + br) * K + (k0 + bc);
        #pragma unroll
        for (int u = 0; u < 8; u += 4) {
            float4 t = bval ? *(const float4*)(bsrc + u) : make_float4(0.f, 0.f, 0.f, 0.f);
            breg[u + 0] = t.x; breg[u + 1] = t.y; breg[u + 2] = t.z; breg[u + 3] = t.w;
        }
    };
    auto stage_store = [&](int buf) {
        _Float16* ad = &As[buf][ar * LDA + ac];
        v8h h0, h1;
        #pragma unroll
        for (int u = 0; u < 8; ++u) { h0[u] = (_Float16)areg[u]; h1[u] = (_Float16)areg[8 + u]; }
        *(v8h*)(ad)     = h0;
        *(v8h*)(ad + 8) = h1;
        _Float16* bd = &Bs[buf][br * LDB + bc];
        v8h hb;
        #pragma unroll
        for (int u = 0; u < 8; ++u) hb[u] = (_Float16)breg[u];
        *(v8h*)bd = hb;
    };
    // A frag: lanes 0-15 -> M=mn; elems 0-7: K=g*8+j, elems 8-15: K=16+g*8+j
    auto frag_a = [&](const _Float16* base, int msub) -> v16h {
        const _Float16* ap = base + (wm * 32 + msub + mn) * LDA;
        v8h lo = *(const v8h*)(ap + g * 8);
        v8h hi = *(const v8h*)(ap + 16 + g * 8);
        v16h f;
        #pragma unroll
        for (int u = 0; u < 8; ++u) { f[u] = lo[u]; f[u + 8] = hi[u]; }
        return f;
    };
    // B frag ([n][k] LDS layout): lane -> N=mn; elems 0-7: K=g*8+j, elems 8-15: K=16+g*8+j
    auto frag_b = [&](const _Float16* base, int nsub) -> v16h {
        const _Float16* bp = base + (wn * 32 + nsub + mn) * LDB;
        v8h lo = *(const v8h*)(bp + g * 8);
        v8h hi = *(const v8h*)(bp + 16 + g * 8);
        v16h f;
        #pragma unroll
        for (int u = 0; u < 8; ++u) { f[u] = lo[u]; f[u + 8] = hi[u]; }
        return f;
    };

    v8f acc00 = {}, acc01 = {}, acc10 = {}, acc11 = {};

    load_regs(0);
    stage_store(0);
    int pbuf = 0;

    for (int k0 = 0; k0 < K; k0 += 32) {
        const bool has_next = (k0 + 32) < K;
        if (has_next) load_regs(k0 + 32);      // global loads in flight over compute
        __syncthreads();                        // buffer pbuf ready

        const _Float16* ab = As[pbuf];
        const _Float16* bb = Bs[pbuf];
        v16h a0 = frag_a(ab, 0);
        v16h a1 = frag_a(ab, 16);
        v16h b0 = frag_b(bb, 0);
        v16h b1 = frag_b(bb, 16);
        acc00 = WMMA_F16(a0, b0, acc00);
        acc01 = WMMA_F16(a0, b1, acc01);
        acc10 = WMMA_F16(a1, b0, acc10);
        acc11 = WMMA_F16(a1, b1, acc11);

        if (has_next) { stage_store(pbuf ^ 1); pbuf ^= 1; }
    }

    // D layout: VGPR v -> M = v + 8*g, N = mn
    #pragma unroll
    for (int j = 0; j < 2; ++j) {
        const int col = n0 + wn * 32 + j * 16 + mn;
        if (col >= N) continue;
        const float bv = bias ? bias[col] : 0.f;
        #pragma unroll
        for (int i = 0; i < 2; ++i) {
            const v8f* accp = (i == 0) ? (j == 0 ? &acc00 : &acc01)
                                       : (j == 0 ? &acc10 : &acc11);
            #pragma unroll
            for (int v = 0; v < 8; ++v) {
                const int rowm = m0 + wm * 32 + i * 16 + g * 8 + v;
                if (rowm < M) C[(size_t)rowm * N + col] = (*accp)[v] + bv;
            }
        }
    }
}

// ---------------------------------------------------------------------------
// Deformable sampling: one wave per (b,s,h); lane = head channel (HD==32).
// Per-wave logits are staged global->LDS with GLOBAL_LOAD_ASYNC_TO_LDS_B128
// (ASYNCcnt-tracked, no VGPR round trip), then consumed after
// s_wait_asynccnt 0 + block barrier. Gathers from `values` are 32
// consecutive floats -> one coalesced 128B txn per corner, served from L2.
// ---------------------------------------------------------------------------
#define DB 4
#define DS 1024
#define DH 8
#define DL 4
#define DP 20
#define DT 20197
#define OFF_SCALE 4.0f

__global__ void __launch_bounds__(256)
deform_sample(const float* __restrict__ values,    // (B, T, 256)
              const float* __restrict__ off_log,   // (B*S, 1280)
              const float* __restrict__ attn_log,  // (B*S, 640)
              const float* __restrict__ ref_log,   // (B*S, 32)
              const float* __restrict__ def_refs,  // (4, 2)
              float* __restrict__ out_mid)         // (B*S, 256)
{
    constexpr int HH[DL]    = {100, 50, 25, 13};
    constexpr int WW[DL]    = {152, 76, 38, 19};
    constexpr int START[DL] = {0, 15200, 19000, 19950};

    __shared__ __align__(16) float s_attn[8][80];
    __shared__ __align__(16) float s_off [8][160];
    __shared__ __align__(16) float s_ref [8][4];

    const int wave = threadIdx.x >> 5;
    const int lane = threadIdx.x & 31;
    const int idx  = blockIdx.x * 8 + wave;   // (b*S+s)*H + h
    const int h    = idx & 7;
    const int row  = idx >> 3;                // b*S + s
    const int b    = row / DS;

    // ---- async stage this wave's logits into LDS (b128 per lane) ----
    {
        const float* ga = attn_log + (size_t)row * (DH * DL * DP) + h * (DL * DP);
        const unsigned la = (unsigned)(size_t)(void*)&s_attn[wave][0];
        if (lane < 20) {   // 80 floats = 20 x b128
            const float* g = ga + lane * 4;
            const unsigned l = la + lane * 16;
            asm volatile("global_load_async_to_lds_b128 %0, %1, off"
                         :: "v"(l), "v"(g) : "memory");
        }
        const float* go = off_log + (size_t)row * (DH * DL * DP * 2) + h * (DL * DP * 2);
        const unsigned lo = (unsigned)(size_t)(void*)&s_off[wave][0];
        {   // 160 floats = 40 x b128 : 32 lanes + 8 lanes
            const float* g = go + lane * 4;
            const unsigned l = lo + lane * 16;
            asm volatile("global_load_async_to_lds_b128 %0, %1, off"
                         :: "v"(l), "v"(g) : "memory");
        }
        if (lane < 8) {
            const float* g = go + (32 + lane) * 4;
            const unsigned l = lo + (32 + lane) * 16;
            asm volatile("global_load_async_to_lds_b128 %0, %1, off"
                         :: "v"(l), "v"(g) : "memory");
        }
        if (lane == 0) {   // 4 ref logits = 1 x b128
            const float* g = ref_log + (size_t)row * 32 + h * 4;
            const unsigned l = (unsigned)(size_t)(void*)&s_ref[wave][0];
            asm volatile("global_load_async_to_lds_b128 %0, %1, off"
                         :: "v"(l), "v"(g) : "memory");
        }
    }
    asm volatile("s_wait_asynccnt 0x0" ::: "memory");
    __syncthreads();

    // Reference-point softmax over R=4 mixed with sigmoid(def_refs)
    float rm = fmaxf(fmaxf(s_ref[wave][0], s_ref[wave][1]),
                     fmaxf(s_ref[wave][2], s_ref[wave][3]));
    float ax = 0.f, ay = 0.f, rs = 0.f;
    #pragma unroll
    for (int r = 0; r < 4; ++r) {
        float e = __expf(s_ref[wave][r] - rm);
        rs += e;
        ax += e * (1.f / (1.f + __expf(-def_refs[r * 2 + 0])));
        ay += e * (1.f / (1.f + __expf(-def_refs[r * 2 + 1])));
    }
    ax /= rs; ay /= rs;

    // Attention softmax denominator over the 80 (level,point) logits
    float am = -1e30f;
    for (int j = 0; j < DL * DP; ++j) am = fmaxf(am, s_attn[wave][j]);
    float as = 0.f;
    for (int j = 0; j < DL * DP; ++j) as += __expf(s_attn[wave][j] - am);
    const float inv_as = 1.f / as;

    const int c = lane;                       // channel within head
    float acc = 0.f;

    #pragma unroll
    for (int lvl = 0; lvl < DL; ++lvl) {
        const int hh = HH[lvl], ww = WW[lvl];
        const float* img = values + ((size_t)b * DT + START[lvl]) * 256 + h * 32 + c;
        const float sx = OFF_SCALE / (float)ww;
        const float sy = OFF_SCALE / (float)hh;
        for (int p = 0; p < DP; ++p) {
            const int lp = lvl * DP + p;
            const float w = __expf(s_attn[wave][lp] - am) * inv_as;
            float lx = ax + tanhf(s_off[wave][lp * 2 + 0]) * sx;
            float ly = ay + tanhf(s_off[wave][lp * 2 + 1]) * sy;
            lx = fminf(fmaxf(lx, 0.f), 1.f);
            ly = fminf(fmaxf(ly, 0.f), 1.f);
            // grid = 2*clip-1; pixel x = ((gx+1)*W - 1)/2 = clip*W - 0.5
            const float xf = lx * (float)ww - 0.5f;
            const float yf = ly * (float)hh - 0.5f;
            const float x0 = floorf(xf), y0 = floorf(yf);
            const float wx = xf - x0, wy = yf - y0;
            const int xi = (int)x0, yi = (int)y0;
            const bool xv0 = (xi >= 0) & (xi < ww);
            const bool xv1 = (xi + 1 >= 0) & (xi + 1 < ww);
            const bool yv0 = (yi >= 0) & (yi < hh);
            const bool yv1 = (yi + 1 >= 0) & (yi + 1 < hh);
            float v00 = 0.f, v01 = 0.f, v10 = 0.f, v11 = 0.f;
            if (yv0) {
                const float* rp = img + (size_t)(yi * ww) * 256;
                if (xv0) v00 = rp[(size_t)xi * 256];
                if (xv1) v01 = rp[(size_t)(xi + 1) * 256];
            }
            if (yv1) {
                const float* rp = img + (size_t)((yi + 1) * ww) * 256;
                if (xv0) v10 = rp[(size_t)xi * 256];
                if (xv1) v11 = rp[(size_t)(xi + 1) * 256];
            }
            const float bil = v00 * (1.f - wx) * (1.f - wy) + v01 * wx * (1.f - wy)
                            + v10 * (1.f - wx) * wy         + v11 * wx * wy;
            acc += w * bil;
        }
    }
    out_mid[(size_t)row * 256 + h * 32 + c] = acc;
}

// ---------------------------------------------------------------------------
extern "C" void kernel_launch(void* const* d_in, const int* in_sizes, int n_in,
                              void* d_out, int out_size, void* d_ws, size_t ws_size,
                              hipStream_t stream) {
    const float* x        = (const float*)d_in[0];   // (4,1024,256)
    const float* enc      = (const float*)d_in[1];   // (4,20197,256)
    const float* v_w      = (const float*)d_in[2];   // (256,256)
    const float* out_w    = (const float*)d_in[3];   // (256,256)
    const float* off_w    = (const float*)d_in[4];   // (1280,256)
    const float* off_b    = (const float*)d_in[5];   // (1280)
    const float* attn_w   = (const float*)d_in[6];   // (640,256)
    const float* attn_b   = (const float*)d_in[7];   // (640)
    const float* ref_w    = (const float*)d_in[8];   // (32,256)
    const float* ref_b    = (const float*)d_in[9];   // (32)
    const float* def_refs = (const float*)d_in[10];  // (4,2)
    float* out = (float*)d_out;                      // (4096,256)

    const int MV = 4 * 20197;   // 80788 rows of value projection
    const int MX = 4 * 1024;    // 4096 query rows
    const int K  = 256;

    // Workspace carve-up (floats)
    float* values   = (float*)d_ws;                         // 80788*256
    float* off_log  = values   + (size_t)MV * 256;          // 4096*1280
    float* attn_log = off_log  + (size_t)MX * 1280;         // 4096*640
    float* ref_log  = attn_log + (size_t)MX * 640;          // 4096*32
    float* out_mid  = ref_log  + (size_t)MX * 32;           // 4096*256

    const dim3 blk(256);

    // 1) values = enc @ v_w.T            (80788 x 256)
    gemm_f16wmma_bias<<<dim3((MV + 127) / 128, (256 + 63) / 64), blk, 0, stream>>>(
        enc, v_w, nullptr, values, MV, 256, K);
    // 2) offset logits = x @ off_w.T + b (4096 x 1280)
    gemm_f16wmma_bias<<<dim3(MX / 128, 1280 / 64), blk, 0, stream>>>(
        x, off_w, off_b, off_log, MX, 1280, K);
    // 3) attn logits = x @ attn_w.T + b  (4096 x 640)
    gemm_f16wmma_bias<<<dim3(MX / 128, 640 / 64), blk, 0, stream>>>(
        x, attn_w, attn_b, attn_log, MX, 640, K);
    // 4) ref logits = x @ ref_w.T + b    (4096 x 32, N guarded)
    gemm_f16wmma_bias<<<dim3(MX / 128, 1), blk, 0, stream>>>(
        x, ref_w, ref_b, ref_log, MX, 32, K);
    // 5) deformable sampling -> out_mid  (one wave per (b,s,h))
    deform_sample<<<dim3((4 * 1024 * 8) / 8), blk, 0, stream>>>(
        values, off_log, attn_log, ref_log, def_refs, out_mid);
    // 6) out = out_mid @ out_w.T         (4096 x 256)
    gemm_f16wmma_bias<<<dim3(MX / 128, 256 / 64), blk, 0, stream>>>(
        out_mid, out_w, nullptr, out, MX, 256, K);
}
